// ContigSearch_3118146257641
// MI455X (gfx1250) — compile-verified, matrix operand order checked by reading, not run on hardware
//
#include <hip/hip_runtime.h>
#include <hip/hip_bf16.h>
#include <math.h>

// ---------------- problem constants ----------------
#define LL   256
#define CCH  138
#define CPAD 160          // channels padded to multiple of 32 for WMMA K-chunks
#define EPSV 1e-8f

// TDM staging: section = 6 q's worth of Toeplitz B = 6*5120 B
#define SEC_Q      6
#define SEC_BYTES  30720
#define SEC_HALF   15360

typedef _Float16 v16h __attribute__((ext_vector_type(16)));
typedef _Float16 v8h  __attribute__((ext_vector_type(8)));
typedef float    v8f  __attribute__((ext_vector_type(8)));

// contig tables (compile-time facts of the reference)
__constant__ int dCS[4]  = {5,30,55,90};
__constant__ int dLEN[4] = {15,12,20,15};
// off-diagonal pairs p=0..5 : (0,1)(0,2)(0,3)(1,2)(1,3)(2,3)
__constant__ int dPI[6]  = {0,0,0,1,1,2};
__constant__ int dPJ[6]  = {1,2,3,2,3,3};
__constant__ int dNI[6]  = {15,15,15,12,12,20};
__constant__ int dNJ[6]  = {12,20,15,20,15,15};
__constant__ int dW [6]  = {27,35,30,35,30,30};       // window = nj+15
__constant__ int dNQ[6]  = {405,525,450,420,360,600}; // ni*W
__constant__ int dBTZ[7] = {0,1036800,2380800,3532800,4608000,5529600,7065600}; // half offsets
__constant__ int dS1[6]  = {5,5,5,30,30,55};
__constant__ int dS2[6]  = {30,55,90,55,90,90};
// mask: keep iff y,x in [LO,HI) and x-y >= THR
__constant__ int dLO[6]  = {0,0,0,15,15,27};
__constant__ int dHI[6]  = {210,222,242,222,242,242};
__constant__ int dTHR[6] = {15,27,47,12,32,20};
// diag job prefix: n_i = 257-len
__constant__ int dNPFX[5] = {0,242,487,724,966};

// d_out element offsets (all f32)
#define OFF_CCE2D 0
#define OFF_CCE1D 1048576
#define OFF_KL2D  1049600
#define OFF_KL1D  2098176
#define OFF_KLTOT 2099200
#define OFF_IJ2D  2099201
#define OFF_IJ1D  3147777
#define OFF_IJTOT 3148801

__device__ inline bool is_zero_ch(int c){
    return c==0 || c==25 || c==38 || c==75 || c==100 || c==125;
}

// ---------------- TDM: async-copy one B section (30720 B, contiguous) into LDS ----------------
// D# per CDNA5 ISA ch8: group0 = {count=1, lds_addr, global_addr, type=2},
// group1 = {mask=0, data_size=8B, tensor_dim0=16M, tensor_dim1=1, tile_dim0=3840,
//           tile_dim1=1, dim0_stride=3840}. 1-row tile => straight 30720-byte DMA.
__device__ __forceinline__ void tdm_issue_b(unsigned long long gaddr, unsigned lds_addr)
{
    unsigned g0w0 = 1u;                                            // count=1 (valid), no gather
    unsigned g0w1 = lds_addr;                                      // LDS byte address
    unsigned g0w2 = (unsigned)(gaddr & 0xffffffffu);               // global_addr[31:0]
    unsigned g0w3 = (unsigned)((gaddr >> 32) & 0x01ffffffu) | (2u<<30); // addr[56:32] | type=2
    unsigned g1w0 = 3u << 16;                                      // data_size = 8B
    unsigned g1w1 = 0u;                                            // tensor_dim0[15:0] = 0
    unsigned g1w2 = 0x0100u | (1u << 16);                          // tensor_dim0=0x01000000, tensor_dim1=1
    unsigned g1w3 = 3840u << 16;                                   // tile_dim0 = 3840 (x8B = 30720B)
    unsigned g1w4 = 1u;                                            // tile_dim1 = 1, tile_dim2 = 0
    unsigned g1w5 = 3840u;                                         // tensor_dim0_stride
    unsigned g1w6 = 0u;
    unsigned g1w7 = 0u;
    asm volatile(
        "s_mov_b32 s40, %0\n\t"
        "s_mov_b32 s41, %1\n\t"
        "s_mov_b32 s42, %2\n\t"
        "s_mov_b32 s43, %3\n\t"
        "s_mov_b32 s44, %4\n\t"
        "s_mov_b32 s45, %5\n\t"
        "s_mov_b32 s46, %6\n\t"
        "s_mov_b32 s47, %7\n\t"
        "s_mov_b32 s48, %8\n\t"
        "s_mov_b32 s49, %9\n\t"
        "s_mov_b32 s50, %10\n\t"
        "s_mov_b32 s51, %11\n\t"
        "tensor_load_to_lds s[40:43], s[44:51]\n\t"
        :
        : "s"(g0w0), "s"(g0w1), "s"(g0w2), "s"(g0w3),
          "s"(g1w0), "s"(g1w1), "s"(g1w2), "s"(g1w3),
          "s"(g1w4), "s"(g1w5), "s"(g1w6), "s"(g1w7)
        : "s40","s41","s42","s43","s44","s45","s46","s47","s48","s49","s50","s51","memory");
}

// ---------------- kernel 0: zero accumulators ----------------
__global__ __launch_bounds__(256) void k_init(float* __restrict__ acc, float* __restrict__ kltot)
{
    int id = blockIdx.x*256 + threadIdx.x;
    if (id < 6*65536) acc[id] = 0.f;
    if (id == 0) *kltot = 0.f;
}

// ---------------- kernel 1: neg_log_p (f16, padded) + kl_ij + kl_tot ----------------
__global__ __launch_bounds__(256) void k_prep(const float* __restrict__ hal,
                                              const float* __restrict__ bkg,
                                              _Float16* __restrict__ nlp,
                                              float* __restrict__ klij,
                                              float* __restrict__ kltot)
{
    int pid = blockIdx.x*256 + threadIdx.x;      // 65536 pixels
    int y = pid >> 8, x = pid & 255;
    const float* hp = hal + (size_t)pid*CCH;
    const float* bp = bkg + (size_t)pid*CCH;
    _Float16*    np = nlp + (size_t)pid*CPAD;
    float kl = 0.f;
    for (int c=0; c<CCH; ++c){
        float p = hp[c], b = bp[c];
        np[c] = (_Float16)(-__logf(p + EPSV));
        kl   += -p * __logf(p/(b+EPSV) + EPSV);
    }
    for (int c=CCH; c<CPAD; ++c) np[c] = (_Float16)0.f;
    kl *= (1.f/6.f);
    if (y == x) kl = 0.f;
    klij[pid] = kl;
    atomicAdd(kltot, kl);
}

// ---------------- kernel 2: ker_sizes (sum/6) and raw sums, all 16 (i,j) ----------------
__global__ __launch_bounds__(256) void k_ker(const float* __restrict__ geo,
                                             float* __restrict__ ks, float* __restrict__ kraw)
{
    __shared__ float red[256];
    int b = blockIdx.x; int i = b>>2, j = b&3;
    int ni = dLEN[i], nj = dLEN[j];
    int tot = ni*nj*CCH;
    float s = 0.f;
    for (int t=threadIdx.x; t<tot; t+=256){
        int c = t % CCH; int r = t / CCH;
        int dy = r / nj, dx = r % nj;
        if (!is_zero_ch(c))
            s += geo[((dCS[i]+dy)*128 + dCS[j]+dx)*CCH + c];
    }
    red[threadIdx.x] = s; __syncthreads();
    for (int o=128; o>0; o>>=1){ if (threadIdx.x<o) red[threadIdx.x]+=red[threadIdx.x+o]; __syncthreads(); }
    if (threadIdx.x==0){ kraw[b]=red[0]; ks[b]=red[0]*(1.f/6.f); }
}

// ---------------- kernel 3: channel-summed 2D kernels (for kl convs) ----------------
__global__ __launch_bounds__(256) void k_ker2d(const float* __restrict__ geo, float* __restrict__ k2d)
{
    int id = blockIdx.x*256 + threadIdx.x;       // 10*512
    int pp = id >> 9; int e = id & 511;
    if (pp >= 10) return;
    int i, j;
    if (pp < 6){ i = dPI[pp]; j = dPJ[pp]; } else { i = pp-6; j = pp-6; }
    int ni = dLEN[i], nj = dLEN[j];
    if (e >= ni*nj) return;
    int dy = e / nj, dx = e % nj;
    const float* g = geo + ((dCS[i]+dy)*128 + dCS[j]+dx)*CCH;
    float s = 0.f;
    for (int c=0; c<CCH; ++c) if (!is_zero_ch(c)) s += g[c];
    k2d[pp*512 + e] = s*(1.f/6.f);
}

// ---------------- kernel 4: build pre-swizzled Toeplitz B matrices (f16) ----------------
// layout per pair: [k_chunk][lane(32)][16 halfs] == exactly the per-lane B fragment load
__global__ __launch_bounds__(256) void k_toep(const float* __restrict__ geo, _Float16* __restrict__ btz)
{
    int e = blockIdx.x*256 + threadIdx.x;
    if (e >= 7065600) return;
    int p = 0;
    while (p < 5 && e >= dBTZ[p+1]) ++p;
    int loc    = e - dBTZ[p];
    int chunk  = loc >> 9;
    int within = loc & 511;
    int lane = within >> 4, el = within & 15;
    int base = (lane < 16) ? 0 : 8;
    int klc  = base + el + ((el >= 8) ? 8 : 0);   // hw B-fragment K index within chunk
    int k    = chunk*32 + klc;
    int n    = lane & 15;                          // output column xx
    int W    = dW[p];
    int q    = k / CPAD;  int c = k % CPAD;
    int dy   = q / W,     wr = q % W;
    int dx   = wr - n;
    float v = 0.f;
    if (c < CCH && !is_zero_ch(c) && dx >= 0 && dx < dNJ[p] && dy < dNI[p])
        v = geo[((dS1[p]+dy)*128 + dS2[p]+dx)*CCH + c];
    btz[e] = (_Float16)v;
}

// ---------------- kernel 5: broadcast fills (ij_*, diag inf blocks, 1D zeros) ----------------
__global__ __launch_bounds__(256) void k_fill(float* __restrict__ out, const float* __restrict__ ks)
{
    int idx = blockIdx.x*256 + threadIdx.x;
    const float PINF = __builtin_inff();
    if (idx < (1<<20)){
        int p2 = idx >> 16; int i = p2>>2, j = p2&3;
        out[OFF_IJ2D + idx] = (i==j) ? 0.f : ks[p2];
    }
    if (idx < 65536){
        #pragma unroll
        for (int i=0;i<4;++i){
            out[OFF_CCE2D + ((i*5)<<16) + idx] =  PINF;   // diag blocks fully masked
            out[OFF_KL2D  + ((i*5)<<16) + idx] = -PINF;
        }
    }
    if (idx < 1024){
        out[OFF_CCE1D + idx] = 0.f;
        out[OFF_KL1D  + idx] = 0.f;
        out[OFF_IJ1D  + idx] = ks[(idx>>8)*5];
    }
    if (idx == 0) out[OFF_IJTOT] = (float)(LL*(LL-1));
}

// ---------------- kernel 6: WMMA Toeplitz-GEMM conv, TDM-fed, double-buffered LDS B ----------------
// block = (pair, kslice, xtile); 8 waves x 2 y-tiles each = all 16 y-tiles share one B stream.
__global__ __launch_bounds__(256) void k_wmma(const _Float16* __restrict__ nlp,
                                              const _Float16* __restrict__ btz,
                                              float* __restrict__ accbuf)
{
    __shared__ __align__(128) _Float16 smem[2*SEC_HALF];   // 61440 B, double buffer

    const int tid  = threadIdx.x;
    const int lane = tid & 31;
    const int wv   = tid >> 5;            // 0..7
    int blk = blockIdx.x;                 // 384 = 6p * 4s * 16tx
    const int tx = blk & 15; blk >>= 4;
    const int s  = blk & 3;  blk >>= 2;
    const int p  = blk;

    const int W  = dW[p];
    const int nq = dNQ[p];
    const int x0 = tx*16;
    const int q0 = (nq *  s     ) >> 2;
    const int q1 = (nq * (s + 1)) >> 2;
    const int nsec = (q1 - q0 + SEC_Q - 1) / SEC_Q;

    const int y0a = wv*16, y0b = (wv+8)*16;      // two 16-row tiles per wave
    const int m     = lane & 15;
    const int cbase = (lane < 16) ? 0 : 8;       // A-fragment K sub-offset per hw layout

    v8f accA = {0.f,0.f,0.f,0.f,0.f,0.f,0.f,0.f};
    v8f accB = {0.f,0.f,0.f,0.f,0.f,0.f,0.f,0.f};

    const _Float16* bgl = btz + dBTZ[p] + (size_t)q0*2560;   // this slice's B stream (global)
    const unsigned  lds0 = (unsigned)(size_t)(void*)&smem[0];

    if (tid == 0)                                            // prologue: DMA section 0
        tdm_issue_b((unsigned long long)(size_t)bgl, lds0);

    int dy = q0 / W;
    int wr = q0 - dy*W;
    int q  = q0;

    for (int sec = 0; sec < nsec; ++sec){
        if (tid == 0){
            if (sec + 1 < nsec){                             // DMA next section into other buffer
                tdm_issue_b((unsigned long long)(size_t)(bgl + (size_t)(sec+1)*SEC_Q*2560),
                            lds0 + (unsigned)(((sec+1)&1)*SEC_BYTES));
                __builtin_amdgcn_s_wait_tensorcnt(1);        // current section landed
            } else {
                __builtin_amdgcn_s_wait_tensorcnt(0);
            }
        }
        __syncthreads();                                     // B section visible to all waves

        const _Float16* bsec = smem + (sec&1)*SEC_HALF;
        int qe = q0 + (sec+1)*SEC_Q; if (qe > q1) qe = q1;

        for (; q < qe; ++q){
            int rowa = y0a + m + dy; if (rowa > 255) rowa = 255;   // halo clamp (masked region)
            int rowb = y0b + m + dy; if (rowb > 255) rowb = 255;
            int col  = x0 + wr;      if (col  > 255) col  = 255;
            const _Float16* apA = nlp + ((rowa<<8) + col)*CPAD + cbase;
            const _Float16* apB = nlp + ((rowb<<8) + col)*CPAD + cbase;
            const _Float16* bp  = bsec + (q - (q0 + sec*SEC_Q))*2560 + lane*16;
            __builtin_prefetch(apA + CPAD, 0, 1);                  // next q's A row
            #pragma unroll
            for (int cc = 0; cc < 5; ++cc){                        // 160 ch = 5 chunks of K=32
                v8h b0 = *(const v8h*)(bp);
                v8h b1 = *(const v8h*)(bp + 8);
                v16h bf = __builtin_shufflevector(b0, b1, 0,1,2,3,4,5,6,7,8,9,10,11,12,13,14,15);
                v8h a0 = *(const v8h*)(apA);
                v8h a1 = *(const v8h*)(apA + 16);
                v16h afA = __builtin_shufflevector(a0, a1, 0,1,2,3,4,5,6,7,8,9,10,11,12,13,14,15);
                accA = __builtin_amdgcn_wmma_f32_16x16x32_f16(false, afA, false, bf,
                                                              (short)0, accA, false, false);
                v8h a2 = *(const v8h*)(apB);
                v8h a3 = *(const v8h*)(apB + 16);
                v16h afB = __builtin_shufflevector(a2, a3, 0,1,2,3,4,5,6,7,8,9,10,11,12,13,14,15);
                accB = __builtin_amdgcn_wmma_f32_16x16x32_f16(false, afB, false, bf,
                                                              (short)0, accB, false, false);
                apA += 32; apB += 32; bp += 512;
            }
            ++wr; if (wr == W){ wr = 0; ++dy; }
        }
        __syncthreads();                                     // done with this buffer
    }

    // C/D layout: vgpr r -> row y0 + r + (lane<16?0:8), col x0 + (lane&15)
    float* outp = accbuf + p*65536;
    const int x   = x0 + (lane & 15);
    const int ybA = y0a + ((lane < 16) ? 0 : 8);
    const int ybB = y0b + ((lane < 16) ? 0 : 8);
    #pragma unroll
    for (int r = 0; r < 8; ++r){
        atomicAdd(outp + (((ybA + r) << 8) + x), accA[r]);
        atomicAdd(outp + (((ybB + r) << 8) + x), accB[r]);
    }
}

// ---------------- kernel 7: 1-channel kl conv (off-diagonal pairs) + mask + symmetrize ----------------
__global__ __launch_bounds__(256) void k_klconv(const float* __restrict__ klij,
                                                const float* __restrict__ k2d,
                                                float* __restrict__ out)
{
    int id = blockIdx.x*256 + threadIdx.x;
    int p = id >> 16; if (p >= 6) return;
    int pix = id & 65535; int y = pix>>8, x = pix&255;
    int i = dPI[p], j = dPJ[p];
    bool msk = (y>=dLO[p]) && (y<dHI[p]) && (x>=dLO[p]) && (x<dHI[p]) && ((x - y) >= dTHR[p]);
    float v = -__builtin_inff();
    if (msk){
        int ni = dNI[p], nj = dNJ[p];
        const float* kk = k2d + p*512;
        float sv = 0.f;
        for (int dy=0; dy<ni; ++dy){
            const float* row = klij + ((y+dy)<<8) + x;
            for (int dx=0; dx<nj; ++dx) sv += row[dx]*kk[dy*nj+dx];
        }
        v = sv;
    }
    out[OFF_KL2D + ((i*4+j)<<16) + pix]          = v;
    out[OFF_KL2D + ((j*4+i)<<16) + ((x<<8) + y)] = v;
}

// ---------------- kernel 8: diagonal pairs -> cce_1D / kl_1D (one wave per output) ----------------
__global__ __launch_bounds__(256) void k_diag(const _Float16* __restrict__ nlp,
                                              const float* __restrict__ geo,
                                              const float* __restrict__ klij,
                                              const float* __restrict__ k2d,
                                              const float* __restrict__ kraw,
                                              float* __restrict__ out)
{
    int wave = (blockIdx.x*256 + threadIdx.x) >> 5;
    int lane = threadIdx.x & 31;
    if (wave >= dNPFX[4]) return;
    int i = 0; while (i < 3 && wave >= dNPFX[i+1]) ++i;
    int d  = wave - dNPFX[i];
    int ni = dLEN[i];

    int tot = ni*ni*CCH;
    float s = 0.f;
    for (int t = lane; t < tot; t += 32){
        int c = t % CCH; int r = t / CCH; int dy = r/ni, dx = r%ni;
        if (!is_zero_ch(c))
            s += (float)nlp[(size_t)(((d+dy)<<8) + (d+dx))*CPAD + c]
               * geo[((dCS[i]+dy)*128 + dCS[i]+dx)*CCH + c];
    }
    for (int o=16; o>0; o>>=1) s += __shfl_xor(s, o, 32);

    float sk = 0.f;
    const float* kk = k2d + (6+i)*512;
    for (int t = lane; t < ni*ni; t += 32){
        int dy = t/ni, dx = t%ni;
        sk += klij[((d+dy)<<8) + (d+dx)] * kk[t];
    }
    for (int o=16; o>0; o>>=1) sk += __shfl_xor(sk, o, 32);

    if (lane == 0){
        float raw = kraw[i*5];
        out[OFF_CCE1D + i*256 + d] = (raw < 0.5f) ? 1.0e6f : s*(1.f/6.f);
        out[OFF_KL1D  + i*256 + d] = sk;
    }
}

// ---------------- kernel 9: finalize cce_2D (scale, 1e6 substitution, mask, symmetrize) ----------------
__global__ __launch_bounds__(256) void k_final(const float* __restrict__ acc,
                                               const float* __restrict__ kraw,
                                               const float* __restrict__ kltot,
                                               float* __restrict__ out)
{
    int id = blockIdx.x*256 + threadIdx.x;
    if (id == 0) out[OFF_KLTOT] = *kltot;
    int p = id >> 16; if (p >= 6) return;
    int pix = id & 65535; int y = pix>>8, x = pix&255;
    int i = dPI[p], j = dPJ[p];
    bool msk = (y>=dLO[p]) && (y<dHI[p]) && (x>=dLO[p]) && (x<dHI[p]) && ((x - y) >= dTHR[p]);
    float v = __builtin_inff();
    if (msk){
        float raw = kraw[i*4+j];
        v = (raw < 0.5f) ? 1.0e6f : acc[p*65536 + pix]*(1.f/6.f);
    }
    out[OFF_CCE2D + ((i*4+j)<<16) + pix]          = v;
    out[OFF_CCE2D + ((j*4+i)<<16) + ((x<<8) + y)] = v;
}

// ---------------- host launcher ----------------
extern "C" void kernel_launch(void* const* d_in, const int* in_sizes, int n_in,
                              void* d_out, int out_size, void* d_ws, size_t ws_size,
                              hipStream_t stream)
{
    const float* hal = (const float*)d_in[0];
    const float* bkg = (const float*)d_in[1];
    const float* geo = (const float*)d_in[3];   // beta (d_in[2]) unused in forward
    float* out = (float*)d_out;
    char*  ws  = (char*)d_ws;

    _Float16* nlp   = (_Float16*)(ws + 0);           // 256*256*160 halfs = 20,971,520 B
    float*    klij  = (float*)   (ws + 20971520);    // 65536 f32
    _Float16* btz   = (_Float16*)(ws + 21233664);    // 7,065,600 halfs (+30720 B TDM pad)
    float*    acc   = (float*)   (ws + 35395584);    // 6*65536 f32
    float*    ks    = (float*)   (ws + 36968448);    // 16 f32
    float*    kraw  = (float*)   (ws + 36968512);    // 16 f32
    float*    k2d   = (float*)   (ws + 36968576);    // 10*512 f32
    float*    kltot = (float*)   (ws + 36989056);    // 1 f32

    k_init  <<<1536, 256, 0, stream>>>(acc, kltot);
    k_prep  <<<256,  256, 0, stream>>>(hal, bkg, nlp, klij, kltot);
    k_ker   <<<16,   256, 0, stream>>>(geo, ks, kraw);
    k_ker2d <<<20,   256, 0, stream>>>(geo, k2d);
    k_toep  <<<27600,256, 0, stream>>>(geo, btz);
    k_fill  <<<4096, 256, 0, stream>>>(out, ks);
    k_wmma  <<<384,  256, 0, stream>>>(nlp, btz, acc);
    k_klconv<<<1536, 256, 0, stream>>>(klij, k2d, out);
    k_diag  <<<121,  256, 0, stream>>>(nlp, geo, klij, k2d, kraw, out);
    k_final <<<1536, 256, 0, stream>>>(acc, kraw, kltot, out);
}
